// MultiHeadedAttention_25984552141341
// MI455X (gfx1250) — matrix operand order repaired and checked by reading, not compiled, not run on hardware
//
#include <hip/hip_runtime.h>
#include <hip/hip_bf16.h>

typedef __attribute__((ext_vector_type(16))) _Float16 v16h;
typedef __attribute__((ext_vector_type(8)))  _Float16 h8;
typedef __attribute__((ext_vector_type(4)))  _Float16 h4;
typedef __attribute__((ext_vector_type(4)))  float    f4;
typedef __attribute__((ext_vector_type(8)))  float    v8f;

#define BQ 512
#define LSEQ 71
#define DMODEL 512
#define NH 8
#define DH 64
#define LP 96              // L padded to 6 WMMA tiles
#define SF_FF 128
#define MTOT (BQ * LSEQ)   // 36352 = 284 * 128

// ---------------- WMMA helpers (wave32, CDNA5 layouts, ISA 7.12.2) ----------------

__device__ __forceinline__ v8f wmma_f16(v16h a, v16h b, v8f c) {
  return __builtin_amdgcn_wmma_f32_16x16x32_f16(false, a, false, b, (short)0, c, false, false);
}

// A fragment 16x32 f16 from row-major LDS tile (lda in halfs).
__device__ __forceinline__ v16h ldsA_f16(const _Float16* A, int lda) {
  int lane = threadIdx.x & 31;
  int r = lane & 15, hf = lane >> 4;
  const _Float16* p = A + r * lda + hf * 8;
  v16h a;
#pragma unroll
  for (int i = 0; i < 8; ++i) { a[i] = p[i]; a[i + 8] = p[i + 16]; }
  return a;
}

// Same but source is f32 LDS (convert on load).
__device__ __forceinline__ v16h ldsA_f32(const float* A, int lda) {
  int lane = threadIdx.x & 31;
  int r = lane & 15, hf = lane >> 4;
  const float* p = A + r * lda + hf * 8;
  v16h a;
#pragma unroll
  for (int i = 0; i < 8; ++i) { a[i] = (_Float16)p[i]; a[i + 8] = (_Float16)p[i + 16]; }
  return a;
}

// B fragment 32x16 f16 from LDS holding B^T row-major: Bt[n][k].
__device__ __forceinline__ v16h ldsB_f16(const _Float16* Bt, int ldb) {
  int lane = threadIdx.x & 31;
  int n = lane & 15, hf = lane >> 4;
  const _Float16* p = Bt + n * ldb + hf * 8;
  v16h b;
#pragma unroll
  for (int i = 0; i < 8; ++i) { b[i] = p[i]; b[i + 8] = p[i + 16]; }
  return b;
}

__device__ __forceinline__ float gelu_exact(float x) {
  return 0.5f * x * (1.0f + erff(x * 0.70710678118654752f));
}

// ---------------- Kernel 0a: big-weight transpose (f32 [k][n] -> f16 Wt[n][k]) ----

__global__ __launch_bounds__(256) void transpose_w_kernel(
    const float* __restrict__ Wq, const float* __restrict__ Wk,
    const float* __restrict__ Wv, const float* __restrict__ Wo,
    _Float16* __restrict__ wt) {
  __shared__ _Float16 Ls[64 * 72];
  const int w = blockIdx.z;
  const float* W = (w == 0) ? Wq : (w == 1) ? Wk : (w == 2) ? Wv : Wo;
  _Float16* Wt = wt + (size_t)w * DMODEL * DMODEL;
  const int k0 = blockIdx.x * 64, n0 = blockIdx.y * 64;
  const int tid = threadIdx.x;
  for (int i = tid; i < 64 * 64; i += 256) {
    int kk = i >> 6, n = i & 63;   // coalesced over n
    Ls[kk * 72 + n] = (_Float16)W[(size_t)(k0 + kk) * DMODEL + n0 + n];
  }
  __syncthreads();
  for (int i = tid; i < 64 * 64; i += 256) {
    int n = i >> 6, kk = i & 63;   // coalesced over k
    Wt[(size_t)(n0 + n) * DMODEL + k0 + kk] = Ls[kk * 72 + n];
  }
}

// ---------------- Kernel 0b: softmax-MLP weight prep (transposed, padded f16) -----

__global__ __launch_bounds__(256) void prep_sf_kernel(
    const float* __restrict__ w1, const float* __restrict__ w2,
    _Float16* __restrict__ w1t, _Float16* __restrict__ w2t) {
  const int tid = threadIdx.x;
  for (int i = tid; i < SF_FF * LP; i += 256) {       // w1t[n][k], zero for k>=71
    int n = i / LP, k = i % LP;
    w1t[i] = (k < LSEQ) ? (_Float16)w1[k * SF_FF + n] : (_Float16)0.f;
  }
  for (int i = tid; i < LP * SF_FF; i += 256) {       // w2t[n][k], zero for n>=71
    int n = i / SF_FF, k = i % SF_FF;
    w2t[i] = (n < LSEQ) ? (_Float16)w2[k * LSEQ + n] : (_Float16)0.f;
  }
}

// ---------------- Kernel 1: fused QKV projection, 128x64x32 tiles -----------------

__global__ __launch_bounds__(256) void qkv_proj_kernel(
    const float* __restrict__ Xq, const float* __restrict__ Xk, const float* __restrict__ Xv,
    const _Float16* __restrict__ wt,
    const float* __restrict__ bq, const float* __restrict__ bk, const float* __restrict__ bv,
    _Float16* __restrict__ qws, _Float16* __restrict__ kws, _Float16* __restrict__ vtws) {
  __shared__ _Float16 As[128 * 32];   // lda = 32
  __shared__ _Float16 Bs[64 * 32];    // B^T tile: Bs[n][k]

  const int which = blockIdx.z;
  const float* X = (which == 0) ? Xq : (which == 1) ? Xk : Xv;
  const _Float16* Wt = wt + (size_t)which * DMODEL * DMODEL;
  const float* bias = (which == 0) ? bq : (which == 1) ? bk : bv;

  const int m0 = blockIdx.x * 128;
  const int n0 = blockIdx.y * 64;
  const int tid = threadIdx.x;
  const int wave = tid >> 5, lane = tid & 31;
  const int wm = wave >> 1, wn = wave & 1;   // wave: 32 rows x 32 cols (2x2 tiles)

  v8f c00 = {0.f,0.f,0.f,0.f,0.f,0.f,0.f,0.f}, c01 = c00, c10 = c00, c11 = c00;

  // A-stage register indices: 1024 f4 units, 4 per thread
  int ar[4], ac[4];
#pragma unroll
  for (int it = 0; it < 4; ++it) {
    int u = tid + it * 256;
    ar[it] = u >> 3; ac[it] = (u & 7) * 4;
  }
  const int bn = tid >> 2, bk8 = (tid & 3) * 8;   // 256 h8 units, 1 per thread

  for (int k0 = 0; k0 < DMODEL; k0 += 32) {
    f4 av[4];
#pragma unroll
    for (int it = 0; it < 4; ++it)
      av[it] = *(const f4*)(X + (size_t)(m0 + ar[it]) * DMODEL + k0 + ac[it]);
    h8 bv8 = *(const h8*)(Wt + (size_t)(n0 + bn) * DMODEL + k0 + bk8);
    if (k0 + 32 < DMODEL) {
      __builtin_prefetch(X + (size_t)(m0 + ar[0]) * DMODEL + k0 + 32, 0, 1);
      __builtin_prefetch(Wt + (size_t)(n0 + bn) * DMODEL + k0 + 32, 0, 1);
    }
    __syncthreads();   // previous iteration's WMMA reads done
#pragma unroll
    for (int it = 0; it < 4; ++it) {
      h4 hv = { (_Float16)av[it][0], (_Float16)av[it][1],
                (_Float16)av[it][2], (_Float16)av[it][3] };
      *(h4*)(As + ar[it] * 32 + ac[it]) = hv;
    }
    *(h8*)(Bs + bn * 32 + bk8) = bv8;
    __syncthreads();
    v16h a0 = ldsA_f16(As + (wm * 32) * 32, 32);
    v16h a1 = ldsA_f16(As + (wm * 32 + 16) * 32, 32);
    v16h b0 = ldsB_f16(Bs + (wn * 32) * 32, 32);
    v16h b1 = ldsB_f16(Bs + (wn * 32 + 16) * 32, 32);
    c00 = wmma_f16(a0, b0, c00); c01 = wmma_f16(a0, b1, c01);
    c10 = wmma_f16(a1, b0, c10); c11 = wmma_f16(a1, b1, c11);
  }

  const int r = lane & 15, hf = lane >> 4;
#pragma unroll
  for (int mi = 0; mi < 2; ++mi) {
#pragma unroll
    for (int ni = 0; ni < 2; ++ni) {
      v8f c = (mi == 0) ? (ni == 0 ? c00 : c01) : (ni == 0 ? c10 : c11);
      int n = n0 + wn * 32 + ni * 16 + r;
      int hh = n >> 6, dd = n & 63;
      float bb = bias[n];
#pragma unroll
      for (int j = 0; j < 8; ++j) {
        int m = m0 + wm * 32 + mi * 16 + j + 8 * hf;
        int bi = m / LSEQ, l = m % LSEQ;
        _Float16 v = (_Float16)(c[j] + bb);
        if (which == 0)
          qws[(((size_t)bi * NH + hh) * LP + l) * DH + dd] = v;
        else if (which == 1)
          kws[(((size_t)bi * NH + hh) * LP + l) * DH + dd] = v;
        else
          vtws[(((size_t)bi * NH + hh) * DH + dd) * LP + l] = v;
      }
    }
  }
}

// ---------------- Kernel 2: per-(b,h) attention with score-MLP ---------------------
// Dynamic LDS (110592 B):
//   sc   : float[96][96]                         (36864 B)
//   buf1 : qs[96][64]+ks[96][64] -> w2t[96][128] (24576 B)
//   buf2 : w1t[128][96]          -> ps[96][96]   (24576 B)
//   buf3 : hs[96][128]           -> vt[64][96]   (24576 B)

#define ATTN_SMEM_BYTES (36864 + 3 * 24576)

__global__ __launch_bounds__(256) void attn_kernel(
    const _Float16* __restrict__ qws, const _Float16* __restrict__ kws,
    const _Float16* __restrict__ vtws,
    const _Float16* __restrict__ w1tg, const float* __restrict__ b1,
    const _Float16* __restrict__ w2tg, const float* __restrict__ b2,
    _Float16* __restrict__ attnws) {
  extern __shared__ char smem[];
  float*    sc   = (float*)smem;
  _Float16* buf1 = (_Float16*)(smem + 36864);
  _Float16* buf2 = (_Float16*)(smem + 36864 + 24576);
  _Float16* buf3 = (_Float16*)(smem + 36864 + 49152);

  const int tid = threadIdx.x;
  const int wave = tid >> 5, lane = tid & 31;
  const int r = lane & 15, hf = lane >> 4;
  const size_t bh = blockIdx.x;
  const int bidx = (int)(bh >> 3), hd = (int)(bh & 7);

  _Float16* qs  = buf1;
  _Float16* ksb = buf1 + LP * DH;
  _Float16* w1t = buf2;
  _Float16* hs  = buf3;

  // ---- phase 0: stage Q, K (b128 copies; padded rows carry inert finite data)
  const h8* qg = (const h8*)(qws + bh * (size_t)(LP * DH));
  const h8* kg = (const h8*)(kws + bh * (size_t)(LP * DH));
#pragma unroll
  for (int it = 0; it < 3; ++it) {
    int u = tid + it * 256;          // 768 h8 units
    ((h8*)qs)[u]  = qg[u];
    ((h8*)ksb)[u] = kg[u];
  }
#pragma unroll
  for (int it = 0; it < 6; ++it)     // 1536 h8 units of w1t
    ((h8*)w1t)[tid + it * 256] = ((const h8*)w1tg)[tid + it * 256];
  __syncthreads();

  // ---- phase 1: scores = (Q K^T) / 8
  for (int t = wave; t < 36; t += 8) {
    int mt = t / 6, nt = t % 6;
    v8f c = {0.f,0.f,0.f,0.f,0.f,0.f,0.f,0.f};
#pragma unroll
    for (int kk = 0; kk < DH; kk += 32) {
      v16h a = ldsA_f16(qs + mt * 16 * DH + kk, DH);
      v16h b = ldsB_f16(ksb + nt * 16 * DH + kk, DH);
      c = wmma_f16(a, b, c);
    }
#pragma unroll
    for (int j = 0; j < 8; ++j)
      sc[(mt * 16 + j + 8 * hf) * LP + nt * 16 + r] = c[j] * 0.125f;
  }
  __syncthreads();

  // ---- stage w2t into buf1 (qs/ks dead); phase 2: h = gelu(sc @ w1 + b1)
  _Float16* w2t = buf1;
#pragma unroll
  for (int it = 0; it < 6; ++it)
    ((h8*)w2t)[tid + it * 256] = ((const h8*)w2tg)[tid + it * 256];
  for (int t = wave; t < 48; t += 8) {
    int mt = t / 8, nt = t % 8;
    v8f c = {0.f,0.f,0.f,0.f,0.f,0.f,0.f,0.f};
#pragma unroll
    for (int kk = 0; kk < LP; kk += 32) {
      v16h a = ldsA_f32(sc + mt * 16 * LP + kk, LP);
      v16h b = ldsB_f16(w1t + nt * 16 * LP + kk, LP);
      c = wmma_f16(a, b, c);
    }
#pragma unroll
    for (int j = 0; j < 8; ++j) {
      int m = mt * 16 + j + 8 * hf, n = nt * 16 + r;
      hs[m * SF_FF + n] = (_Float16)gelu_exact(c[j] + b1[n]);
    }
  }
  __syncthreads();

  // ---- phase 3: sc += h @ w2 + b2
  for (int t = wave; t < 36; t += 8) {
    int mt = t / 6, nt = t % 6;
    v8f c = {0.f,0.f,0.f,0.f,0.f,0.f,0.f,0.f};
#pragma unroll
    for (int kk = 0; kk < SF_FF; kk += 32) {
      v16h a = ldsA_f16(hs + mt * 16 * SF_FF + kk, SF_FF);
      v16h b = ldsB_f16(w2t + nt * 16 * SF_FF + kk, SF_FF);
      c = wmma_f16(a, b, c);
    }
#pragma unroll
    for (int j = 0; j < 8; ++j) {
      int m = mt * 16 + j + 8 * hf, n = nt * 16 + r;
      float bb = (n < LSEQ) ? b2[n] : 0.f;
      sc[m * LP + n] += c[j] + bb;
    }
  }
  __syncthreads();

  // ---- phase 4/5: stage V^T (hs dead); softmax rows -> ps
  _Float16* vt = buf3;
  const h8* vg = (const h8*)(vtws + bh * (size_t)(DH * LP));
#pragma unroll
  for (int it = 0; it < 3; ++it)
    ((h8*)vt)[tid + it * 256] = vg[tid + it * 256];
  _Float16* ps = buf2;
  if (tid < LP) {
    float* row = sc + tid * LP;
    float mx = -3.4e38f;
    for (int c = 0; c < LSEQ; ++c) mx = fmaxf(mx, row[c]);
    float s = 0.f;
    for (int c = 0; c < LSEQ; ++c) { float e = __expf(row[c] - mx); row[c] = e; s += e; }
    float inv = 1.0f / s;
    for (int c = 0; c < LP; ++c)
      ps[tid * LP + c] = (c < LSEQ) ? (_Float16)(row[c] * inv) : (_Float16)0.f;
  }
  __syncthreads();

  // ---- phase 6: attn = P @ V
  for (int t = wave; t < 24; t += 8) {
    int mt = t / 4, nt = t % 4;
    v8f c = {0.f,0.f,0.f,0.f,0.f,0.f,0.f,0.f};
#pragma unroll
    for (int kk = 0; kk < LP; kk += 32) {
      v16h a = ldsA_f16(ps + mt * 16 * LP + kk, LP);
      v16h b = ldsB_f16(vt + nt * 16 * LP + kk, LP);
      c = wmma_f16(a, b, c);
    }
#pragma unroll
    for (int j = 0; j < 8; ++j) {
      int l = mt * 16 + j + 8 * hf;
      if (l < LSEQ) {
        int col = hd * DH + nt * 16 + r;
        attnws[((size_t)bidx * LSEQ + l) * DMODEL + col] = (_Float16)c[j];
      }
    }
  }
}

// ---------------- Kernel 3: output projection, 128x64x32 tiles --------------------

__global__ __launch_bounds__(256) void out_proj_kernel(
    const _Float16* __restrict__ A, const _Float16* __restrict__ Wt,
    const float* __restrict__ bo, float* __restrict__ Y) {
  __shared__ _Float16 As[128 * 32];
  __shared__ _Float16 Bs[64 * 32];
  const int m0 = blockIdx.x * 128;
  const int n0 = blockIdx.y * 64;
  const int tid = threadIdx.x;
  const int wave = tid >> 5, lane = tid & 31;
  const int wm = wave >> 1, wn = wave & 1;

  v8f c00 = {0.f,0.f,0.f,0.f,0.f,0.f,0.f,0.f}, c01 = c00, c10 = c00, c11 = c00;

  int ar[2], ac[2];
#pragma unroll
  for (int it = 0; it < 2; ++it) {   // 512 h8 units of A
    int u = tid + it * 256;
    ar[it] = u >> 2; ac[it] = (u & 3) * 8;
  }
  const int bn = tid >> 2, bk8 = (tid & 3) * 8;

  for (int k0 = 0; k0 < DMODEL; k0 += 32) {
    h8 av[2];
#pragma unroll
    for (int it = 0; it < 2; ++it)
      av[it] = *(const h8*)(A + (size_t)(m0 + ar[it]) * DMODEL + k0 + ac[it]);
    h8 bv8 = *(const h8*)(Wt + (size_t)(n0 + bn) * DMODEL + k0 + bk8);
    if (k0 + 32 < DMODEL) {
      __builtin_prefetch(A + (size_t)(m0 + ar[0]) * DMODEL + k0 + 32, 0, 1);
      __builtin_prefetch(Wt + (size_t)(n0 + bn) * DMODEL + k0 + 32, 0, 1);
    }
    __syncthreads();
#pragma unroll
    for (int it = 0; it < 2; ++it)
      *(h8*)(As + ar[it] * 32 + ac[it]) = av[it];
    *(h8*)(Bs + bn * 32 + bk8) = bv8;
    __syncthreads();
    v16h a0 = ldsA_f16(As + (wm * 32) * 32, 32);
    v16h a1 = ldsA_f16(As + (wm * 32 + 16) * 32, 32);
    v16h b0 = ldsB_f16(Bs + (wn * 32) * 32, 32);
    v16h b1 = ldsB_f16(Bs + (wn * 32 + 16) * 32, 32);
    c00 = wmma_f16(a0, b0, c00); c01 = wmma_f16(a0, b1, c01);
    c10 = wmma_f16(a1, b0, c10); c11 = wmma_f16(a1, b1, c11);
  }

  const int r = lane & 15, hf = lane >> 4;
#pragma unroll
  for (int mi = 0; mi < 2; ++mi) {
#pragma unroll
    for (int ni = 0; ni < 2; ++ni) {
      v8f c = (mi == 0) ? (ni == 0 ? c00 : c01) : (ni == 0 ? c10 : c11);
      int n = n0 + wn * 32 + ni * 16 + r;
      float bb = bo[n];
#pragma unroll
      for (int j = 0; j < 8; ++j) {
        int m = m0 + wm * 32 + mi * 16 + j + 8 * hf;
        Y[(size_t)m * DMODEL + n] = c[j] + bb;
      }
    }
  }
}

// ---------------- launch ----------------------------------------------------------

extern "C" void kernel_launch(void* const* d_in, const int* in_sizes, int n_in,
                              void* d_out, int out_size, void* d_ws, size_t ws_size,
                              hipStream_t stream) {
  const float* query = (const float*)d_in[0];
  const float* key   = (const float*)d_in[1];
  const float* value = (const float*)d_in[2];
  const float* Wq = (const float*)d_in[3];
  const float* bq = (const float*)d_in[4];
  const float* Wk = (const float*)d_in[5];
  const float* bk = (const float*)d_in[6];
  const float* Wv = (const float*)d_in[7];
  const float* bv = (const float*)d_in[8];
  const float* Wo = (const float*)d_in[9];
  const float* bo = (const float*)d_in[10];
  const float* sf_w1 = (const float*)d_in[11];
  const float* sf_b1 = (const float*)d_in[12];
  const float* sf_w2 = (const float*)d_in[13];
  const float* sf_b2 = (const float*)d_in[14];

  const size_t QK_HALFS = (size_t)BQ * NH * LP * DH;    // padded per-head Q/K/V^T
  _Float16* qws    = (_Float16*)d_ws;
  _Float16* kws    = qws + QK_HALFS;
  _Float16* vtws   = kws + QK_HALFS;
  _Float16* attnws = vtws + QK_HALFS;                   // [36352][512]
  _Float16* wt     = attnws + (size_t)MTOT * DMODEL;    // 4 x [512][512] f16
  _Float16* w1t    = wt + (size_t)4 * DMODEL * DMODEL;  // [128][96]
  _Float16* w2t    = w1t + SF_FF * LP;                  // [96][128]

  transpose_w_kernel<<<dim3(8, 8, 4), 256, 0, stream>>>(Wq, Wk, Wv, Wo, wt);
  prep_sf_kernel<<<1, 256, 0, stream>>>(sf_w1, sf_w2, w1t, w2t);

  dim3 g1(MTOT / 128, DMODEL / 64, 3);
  qkv_proj_kernel<<<g1, 256, 0, stream>>>(query, key, value, wt, bq, bk, bv,
                                          qws, kws, vtws);

  attn_kernel<<<BQ * NH, 256, ATTN_SMEM_BYTES, stream>>>(
      qws, kws, vtws, w1t, sf_b1, w2t, sf_b2, attnws);

  dim3 g3(MTOT / 128, DMODEL / 64);
  out_proj_kernel<<<g3, 256, 0, stream>>>(attnws, wt + (size_t)3 * DMODEL * DMODEL,
                                          bo, (float*)d_out);
}
